// CHRONOSInference_403726926272
// MI455X (gfx1250) — compile-verified, matrix operand order checked by reading, not run on hardware
//
#include <hip/hip_runtime.h>
#include <hip/hip_bf16.h>
#include <math.h>

// ---------------- problem constants (match reference) ----------------
#define NNODES 20000
#define NEDGE  120000
#define ETOT   (NEDGE + NNODES)   // self-loops appended
#define HEADS  8
#define HID    256
#define NEG_SLOPE 0.2f

// ---------------- WMMA GEMM tile config ----------------
#define BM 64
#define BN 64
#define BK 32
#define LDA_P 40   // padded LDS row stride in bf16 (80B, keeps 16B alignment)
#define LDB_P 40

typedef __attribute__((ext_vector_type(16))) __bf16 v16bf;
typedef __attribute__((ext_vector_type(8)))  __bf16 v8bf;
typedef __attribute__((ext_vector_type(2)))  __bf16 v2bf;
typedef __attribute__((ext_vector_type(8)))  float  v8f;

enum Act { ACT_NONE = 0, ACT_RELU = 1, ACT_ELU = 2 };

__device__ __forceinline__ float act_apply(float v, int act) {
  if (act == ACT_RELU) return v > 0.f ? v : 0.f;
  if (act == ACT_ELU)  return v > 0.f ? v : (expf(v) - 1.f);
  return v;
}

__device__ __forceinline__ int imin(int a, int b) { return a < b ? a : b; }

// Stage one BMxBK A tile and one BKxBN B tile (fp32 -> bf16) into LDS.
// A stored [m][k]; B stored transposed [n][k] with k-pairs packed as b32.
__device__ __forceinline__ void stage_tiles(
    const float* __restrict__ A, const float* __restrict__ B,
    __bf16* __restrict__ As, __bf16* __restrict__ Bs,
    int M, int N, int K, int m0, int n0, int k0,
    int ar, int akc, int bn, int bkp)
{
  const bool fullA = (m0 + BM <= M) && (k0 + BK <= K);
  const bool fullB = (n0 + BN <= N) && (k0 + BK <= K);

  if (fullA) {
    const float* ap = A + (size_t)(m0 + ar) * K + k0 + akc;
    if (k0 + BK < K) __builtin_prefetch(ap + BK, 0, 1);   // global_prefetch_b8
    const float4 f0 = *(const float4*)ap;
    const float4 f1 = *(const float4*)(ap + 4);
    v8bf pk;
    pk[0] = (__bf16)f0.x; pk[1] = (__bf16)f0.y; pk[2] = (__bf16)f0.z; pk[3] = (__bf16)f0.w;
    pk[4] = (__bf16)f1.x; pk[5] = (__bf16)f1.y; pk[6] = (__bf16)f1.z; pk[7] = (__bf16)f1.w;
    *(v8bf*)&As[ar * LDA_P + akc] = pk;
  } else {
    const int gm = m0 + ar;
    const float* ap = A + (size_t)imin(gm, M - 1) * K;
    v8bf pk;
    #pragma unroll
    for (int j = 0; j < 8; ++j) {
      const int gk = k0 + akc + j;
      float v = ap[imin(gk, K - 1)];
      v = (gm < M && gk < K) ? v : 0.f;
      pk[j] = (__bf16)v;
    }
    *(v8bf*)&As[ar * LDA_P + akc] = pk;
  }

  if (fullB) {
    const int gn = n0 + bn;
    #pragma unroll
    for (int q = 0; q < 4; ++q) {
      const int k = 2 * (bkp + q * 4);                    // even k in 0..30
      const float b0 = B[(size_t)(k0 + k)     * N + gn];
      const float b1 = B[(size_t)(k0 + k + 1) * N + gn];
      v2bf p; p[0] = (__bf16)b0; p[1] = (__bf16)b1;
      *(v2bf*)&Bs[bn * LDB_P + k] = p;
    }
  } else {
    const int gn  = n0 + bn;
    const int gnc = imin(gn, N - 1);
    #pragma unroll
    for (int q = 0; q < 4; ++q) {
      const int k = 2 * (bkp + q * 4);
      v2bf p;
      #pragma unroll
      for (int j = 0; j < 2; ++j) {
        const int gk = k0 + k + j;
        float v = B[(size_t)imin(gk, K - 1) * N + gnc];
        v = (gn < N && gk < K) ? v : 0.f;
        p[j] = (__bf16)v;
      }
      *(v2bf*)&Bs[bn * LDB_P + k] = p;
    }
  }
}

// C[M,N] = act(A[M,K] @ B[K,N] + bias), fp32 in/out, bf16 WMMA inner product.
// 256 threads = 8 waves; wave grid 4(M) x 2(N); each wave computes a 16x32 strip.
// LDS double-buffered: next tile staged while current tile feeds the WMMAs;
// one barrier per K-step.
__global__ __launch_bounds__(256)
void gemm_bf16_wmma(const float* __restrict__ A, const float* __restrict__ B,
                    const float* __restrict__ bias, float* __restrict__ C,
                    int M, int N, int K, int act)
{
  __shared__ __bf16 As[2][BM * LDA_P];
  __shared__ __bf16 Bs[2][BN * LDB_P];

  const int tid  = threadIdx.x;
  const int lane = tid & 31;
  const int wid  = tid >> 5;                // 8 waves
  const int m0 = blockIdx.y * BM;
  const int n0 = blockIdx.x * BN;
  const int tm  = (wid >> 1) * 16;          // 0,16,32,48
  const int tn0 = (wid & 1) * 32;           // 0,32

  // staging coordinates (computed once)
  const int ar  = tid >> 2;                 // A: row 0..63
  const int akc = (tid & 3) * 8;            // A: k offset 0,8,16,24
  const int bn  = tid & 63;                 // B: col 0..63
  const int bkp = tid >> 6;                 // B: k-pair base 0..3

  v8f acc0 = {}, acc1 = {};

  stage_tiles(A, B, As[0], Bs[0], M, N, K, m0, n0, 0, ar, akc, bn, bkp);
  __syncthreads();

  int buf = 0;
  for (int k0 = 0; k0 < K; k0 += BK) {
    // stage next tile into the other buffer (overlaps with WMMA below)
    if (k0 + BK < K)
      stage_tiles(A, B, As[buf ^ 1], Bs[buf ^ 1], M, N, K, m0, n0, k0 + BK,
                  ar, akc, bn, bkp);

    // ---- fragments per the 16-bit A-operand lane layout; 2 WMMAs per wave
    {
      const int half = (lane < 16) ? 0 : 8;  // K-base per lane half
      const __bf16* ap  = &As[buf][(tm + (lane & 15)) * LDA_P + half];
      const __bf16* bp0 = &Bs[buf][(tn0 +      (lane & 15)) * LDB_P + half];
      const __bf16* bp1 = &Bs[buf][(tn0 + 16 + (lane & 15)) * LDB_P + half];
      v16bf a, b0, b1;
      #pragma unroll
      for (int i = 0; i < 8; ++i) {
        a[i]  = ap[i];   a[i + 8]  = ap[i + 16];
        b0[i] = bp0[i];  b0[i + 8] = bp0[i + 16];
        b1[i] = bp1[i];  b1[i + 8] = bp1[i + 16];
      }
      acc0 = __builtin_amdgcn_wmma_f32_16x16x32_bf16(
                 false, a, false, b0, (short)0, acc0, false, false);
      acc1 = __builtin_amdgcn_wmma_f32_16x16x32_bf16(
                 false, a, false, b1, (short)0, acc1, false, false);
    }
    __syncthreads();   // protects both: done-reading buf, done-writing buf^1
    buf ^= 1;
  }

  // ---- epilogue: C/D layout — VGPR r: M=r (lanes 0-15) / r+8 (lanes 16-31)
  {
    const int mbase = m0 + tm + ((lane < 16) ? 0 : 8);
    #pragma unroll
    for (int half = 0; half < 2; ++half) {
      const int n = n0 + tn0 + half * 16 + (lane & 15);
      const v8f& acc = half ? acc1 : acc0;
      if (n < N) {
        const float bv = bias ? bias[n] : 0.f;
        #pragma unroll
        for (int r = 0; r < 8; ++r) {
          const int m = mbase + r;
          if (m < M) C[(size_t)m * N + n] = act_apply(acc[r] + bv, act);
        }
      }
    }
  }
}

// ---------------- elementwise / attention kernels ----------------

__global__ void fill_f32(float* __restrict__ p, float v, size_t n) {
  size_t i = (size_t)blockIdx.x * blockDim.x + threadIdx.x;
  if (i < n) p[i] = v;
}

// per (node, head): alpha_src/alpha_dst = <proj[n,h,:], a_src/dst[h,:]>
__global__ void attn_coef(const float* __restrict__ proj,
                          const float* __restrict__ a_src,
                          const float* __restrict__ a_dst,
                          float* __restrict__ as_, float* __restrict__ ad_, int C)
{
  int i = blockIdx.x * blockDim.x + threadIdx.x;
  if (i >= NNODES * HEADS) return;
  const int n = i / HEADS, h = i % HEADS;
  const float* p = proj + (size_t)n * HEADS * C + (size_t)h * C;
  const float* s = a_src + h * C;
  const float* d = a_dst + h * C;
  float vs = 0.f, vd = 0.f;
  for (int c = 0; c < C; ++c) { float x = p[c]; vs += x * s[c]; vd += x * d[c]; }
  as_[i] = vs; ad_[i] = vd;
}

__device__ __forceinline__ void atomicMaxFloat(float* addr, float val) {
  if (val >= 0.f) atomicMax((int*)addr, __float_as_int(val));
  else            atomicMin((unsigned int*)addr, __float_as_uint(val));
}

__device__ __forceinline__ void edge_endpoints(const int* __restrict__ ei,
                                               int e, int& s, int& d) {
  if (e < NEDGE) { s = ei[e]; d = ei[NEDGE + e]; }
  else           { s = e - NEDGE; d = e - NEDGE; }   // self-loop
}

// per (edge, head): e = leaky_relu(as[src]+ad[dst]); segment max into mmax[dst]
__global__ void edge_logits(const int* __restrict__ ei,
                            const float* __restrict__ as_, const float* __restrict__ ad_,
                            float* __restrict__ ee_, float* __restrict__ mx_)
{
  size_t i = (size_t)blockIdx.x * blockDim.x + threadIdx.x;
  if (i >= (size_t)ETOT * HEADS) return;
  const int e = (int)(i / HEADS), h = (int)(i % HEADS);
  int s, d; edge_endpoints(ei, e, s, d);
  float v = as_[s * HEADS + h] + ad_[d * HEADS + h];
  v = (v > 0.f) ? v : NEG_SLOPE * v;
  ee_[i] = v;
  atomicMaxFloat(&mx_[d * HEADS + h], v);
}

// per (edge, head): ex = exp(e - m[dst]); segment sum into den[dst]
__global__ void edge_exp(const int* __restrict__ ei,
                         const float* __restrict__ ee_, const float* __restrict__ mx_,
                         float* __restrict__ ex_, float* __restrict__ dn_)
{
  size_t i = (size_t)blockIdx.x * blockDim.x + threadIdx.x;
  if (i >= (size_t)ETOT * HEADS) return;
  const int e = (int)(i / HEADS), h = (int)(i % HEADS);
  int s, d; edge_endpoints(ei, e, s, d); (void)s;
  const float ex = expf(ee_[i] - mx_[d * HEADS + h]);
  ex_[i] = ex;
  atomicAdd(&dn_[d * HEADS + h], ex);
}

// per (edge, 4-channel chunk of H*C): out[dst] += (ex/den) * proj[src]
__global__ void edge_scatter(const int* __restrict__ ei,
                             const float* __restrict__ ex_, const float* __restrict__ dn_,
                             const float* __restrict__ proj, float* __restrict__ out, int C)
{
  const int chunks = (HEADS * C) >> 2;
  size_t i = (size_t)blockIdx.x * blockDim.x + threadIdx.x;
  if (i >= (size_t)ETOT * chunks) return;
  const int e  = (int)(i / chunks);
  const int ch = (int)(i % chunks);
  const int h  = (ch * 4) / C;
  int s, d; edge_endpoints(ei, e, s, d);
  const float w = ex_[(size_t)e * HEADS + h] / (dn_[d * HEADS + h] + 1e-16f);
  const float4 p = *(const float4*)(proj + (size_t)s * HEADS * C + (size_t)ch * 4);
  float* o = out + (size_t)d * HEADS * C + (size_t)ch * 4;
  atomicAdd(o + 0, w * p.x);
  atomicAdd(o + 1, w * p.y);
  atomicAdd(o + 2, w * p.z);
  atomicAdd(o + 3, w * p.w);
}

__global__ void bias_act_kernel(float* __restrict__ buf, const float* __restrict__ bias,
                                size_t n, int D, int act)
{
  size_t i = (size_t)blockIdx.x * blockDim.x + threadIdx.x;
  if (i >= n) return;
  buf[i] = act_apply(buf[i] + bias[i % (size_t)D], act);
}

// g [N, 8*256] -> out [N, 256], mean over heads
__global__ void head_mean(const float* __restrict__ g, float* __restrict__ out)
{
  size_t i = (size_t)blockIdx.x * blockDim.x + threadIdx.x;
  if (i >= (size_t)NNODES * HID) return;
  const int n = (int)(i / HID), c = (int)(i % HID);
  float s = 0.f;
  #pragma unroll
  for (int h = 0; h < HEADS; ++h) s += g[(size_t)n * (HEADS * HID) + (size_t)h * HID + c];
  out[i] = s * (1.f / HEADS);
}

// comb[n, 0..255] = gmean, comb[n, 256..511] = t
__global__ void concat2(const float* __restrict__ a, const float* __restrict__ b,
                        float* __restrict__ out)
{
  size_t i = (size_t)blockIdx.x * blockDim.x + threadIdx.x;
  if (i >= (size_t)NNODES * 2 * HID) return;
  const int n = (int)(i / (2 * HID)), j = (int)(i % (2 * HID));
  out[i] = (j < HID) ? a[(size_t)n * HID + j] : b[(size_t)n * HID + (j - HID)];
}

// ---------------- host-side orchestration ----------------

static inline dim3 gemm_grid(int M, int N) {
  return dim3((N + BN - 1) / BN, (M + BM - 1) / BM);
}
static inline int blks(size_t n) { return (int)((n + 255) / 256); }

static void run_gat(const float* in, const float* w, const float* asrc, const float* adst,
                    const float* bias, float* proj, float* out, int C, const int* ei,
                    float* as_, float* ad_, float* mx_, float* dn_, float* ex_, float* ee_,
                    hipStream_t stream)
{
  const int D = HEADS * C;
  gemm_bf16_wmma<<<gemm_grid(NNODES, D), 256, 0, stream>>>(in, w, nullptr, proj,
                                                           NNODES, D, HID, ACT_NONE);
  const size_t nh = (size_t)NNODES * HEADS;
  attn_coef<<<blks(nh), 256, 0, stream>>>(proj, asrc, adst, as_, ad_, C);
  fill_f32<<<blks(nh), 256, 0, stream>>>(mx_, -1e30f, nh);
  fill_f32<<<blks(nh), 256, 0, stream>>>(dn_, 0.f, nh);
  const size_t nd = (size_t)NNODES * D;
  fill_f32<<<blks(nd), 256, 0, stream>>>(out, 0.f, nd);
  const size_t eh = (size_t)ETOT * HEADS;
  edge_logits<<<blks(eh), 256, 0, stream>>>(ei, as_, ad_, ee_, mx_);
  edge_exp<<<blks(eh), 256, 0, stream>>>(ei, ee_, mx_, ex_, dn_);
  const size_t tot = (size_t)ETOT * (D / 4);
  edge_scatter<<<blks(tot), 256, 0, stream>>>(ei, ex_, dn_, proj, out, C);
  bias_act_kernel<<<blks(nd), 256, 0, stream>>>(out, bias, nd, D, ACT_ELU);
}

extern "C" void kernel_launch(void* const* d_in, const int* in_sizes, int n_in,
                              void* d_out, int out_size, void* d_ws, size_t ws_size,
                              hipStream_t stream)
{
  (void)in_sizes; (void)n_in; (void)out_size; (void)ws_size;

  // inputs in setup_inputs() order
  const float* x      = (const float*)d_in[0];
  const int*   ei     = (const int*)  d_in[1];
  const float* w_in   = (const float*)d_in[2];
  const float* b_in   = (const float*)d_in[3];
  const float* w_t1   = (const float*)d_in[4];
  const float* b_t1   = (const float*)d_in[5];
  const float* w_t2   = (const float*)d_in[6];
  const float* b_t2   = (const float*)d_in[7];
  const float* w_g1   = (const float*)d_in[8];
  const float* a_src1 = (const float*)d_in[9];
  const float* a_dst1 = (const float*)d_in[10];
  const float* b_g1   = (const float*)d_in[11];
  const float* w_g2   = (const float*)d_in[12];
  const float* a_src2 = (const float*)d_in[13];
  const float* a_dst2 = (const float*)d_in[14];
  const float* b_g2   = (const float*)d_in[15];
  const float* w_g3   = (const float*)d_in[16];
  const float* a_src3 = (const float*)d_in[17];
  const float* a_dst3 = (const float*)d_in[18];
  const float* b_g3   = (const float*)d_in[19];
  const float* w_c1   = (const float*)d_in[20];
  const float* b_c1   = (const float*)d_in[21];
  const float* w_c2   = (const float*)d_in[22];
  const float* b_c2   = (const float*)d_in[23];

  // workspace carve-up (fp32)
  float* ws = (float*)d_ws;
  const size_t n256 = (size_t)NNODES * HID;
  float* h     = ws;             ws += n256;
  float* tmp   = ws;             ws += n256;         // t1 intermediate, then gmean
  float* t     = ws;             ws += n256;
  float* ga    = ws;             ws += n256;         // GAT1 out, then classifier hidden
  float* gb    = ws;             ws += n256;         // GAT2 out
  float* proj  = ws;             ws += (size_t)NNODES * HEADS * HID;  // N x 2048
  float* gout  = ws;             ws += (size_t)NNODES * HEADS * HID;  // N x 2048
  float* comb  = ws;             ws += (size_t)NNODES * 2 * HID;
  float* as_   = ws;             ws += (size_t)NNODES * HEADS;
  float* ad_   = ws;             ws += (size_t)NNODES * HEADS;
  float* mx_   = ws;             ws += (size_t)NNODES * HEADS;
  float* dn_   = ws;             ws += (size_t)NNODES * HEADS;
  float* ex_   = ws;             ws += (size_t)ETOT * HEADS;
  float* ee_   = ws;             ws += (size_t)ETOT * HEADS;

  // h = elu(x @ w_in + b_in)
  gemm_bf16_wmma<<<gemm_grid(NNODES, HID), 256, 0, stream>>>(x, w_in, b_in, h,
                                                             NNODES, HID, 235, ACT_ELU);
  // t = relu(h @ w_t1 + b_t1) @ w_t2 + b_t2
  gemm_bf16_wmma<<<gemm_grid(NNODES, HID), 256, 0, stream>>>(h, w_t1, b_t1, tmp,
                                                             NNODES, HID, HID, ACT_RELU);
  gemm_bf16_wmma<<<gemm_grid(NNODES, HID), 256, 0, stream>>>(tmp, w_t2, b_t2, t,
                                                             NNODES, HID, HID, ACT_NONE);
  // three GAT layers
  run_gat(h,  w_g1, a_src1, a_dst1, b_g1, proj, ga,   HID / HEADS, ei,
          as_, ad_, mx_, dn_, ex_, ee_, stream);
  run_gat(ga, w_g2, a_src2, a_dst2, b_g2, proj, gb,   HID / HEADS, ei,
          as_, ad_, mx_, dn_, ex_, ee_, stream);
  run_gat(gb, w_g3, a_src3, a_dst3, b_g3, proj, gout, HID,         ei,
          as_, ad_, mx_, dn_, ex_, ee_, stream);

  // head mean -> tmp (t1 no longer needed), concat with t
  head_mean<<<blks(n256), 256, 0, stream>>>(gout, tmp);
  concat2<<<blks((size_t)NNODES * 2 * HID), 256, 0, stream>>>(tmp, t, comb);

  // classifier: hidden reuses ga
  gemm_bf16_wmma<<<gemm_grid(NNODES, HID), 256, 0, stream>>>(comb, w_c1, b_c1, ga,
                                                             NNODES, HID, 2 * HID, ACT_RELU);
  gemm_bf16_wmma<<<gemm_grid(NNODES, 2), 256, 0, stream>>>(ga, w_c2, b_c2, (float*)d_out,
                                                           NNODES, 2, HID, ACT_NONE);
}
